// VQVAE_88227218194574
// MI455X (gfx1250) — matrix operand order repaired and checked by reading, not compiled, not run on hardware
//
#include <hip/hip_runtime.h>
#include <hip/hip_bf16.h>
#include <math.h>

// ---------------------------------------------------------------------------
// CDNA5 WMMA types
// ---------------------------------------------------------------------------
typedef __attribute__((ext_vector_type(16))) __bf16 v16bf;
typedef __attribute__((ext_vector_type(8)))  __bf16 v8bf;
typedef __attribute__((ext_vector_type(2)))  __bf16 v2bf;
typedef __attribute__((ext_vector_type(8)))  float  v8f;
typedef __attribute__((ext_vector_type(4)))  unsigned int ui4;
typedef __attribute__((ext_vector_type(8)))  int i8v;
typedef __attribute__((ext_vector_type(4)))  int i4v;

#if __has_builtin(__builtin_amdgcn_tensor_load_to_lds) && __has_builtin(__builtin_amdgcn_s_wait_tensorcnt)
#define USE_TDM 1
#else
#define USE_TDM 0
#endif

__device__ inline v8f wmma_bf16(v16bf a, v16bf b, v8f c) {
  return __builtin_amdgcn_wmma_f32_16x16x32_bf16(
      false, a, false, b, (short)0, c, false, false);
}

#define BM 64
#define BN 64
#define BK 32
#define LDA 40   // bf16 elements per A row (80B, 16B-aligned rows)
#define LDBT 40  // bf16 elements per B row (B stored transposed [N][K])

// A-fragment (16x32 bf16): lane m=lane&15, kh=(lane>>4)*8.
// Per documented layout the lane's 16 values are K = [kh..kh+7] and
// [16+kh..16+kh+7] -> two aligned 16B runs -> 2x ds_load_b128.
__device__ inline v16bf load_a_frag(const __bf16* As, int lane) {
  int m = lane & 15, kh = (lane >> 4) * 8;
  v8bf r0 = *(const v8bf*)(As + m * LDA + kh);
  v8bf r1 = *(const v8bf*)(As + m * LDA + 16 + kh);
  return __builtin_shufflevector(r0, r1, 0, 1, 2, 3, 4, 5, 6, 7,
                                 8, 9, 10, 11, 12, 13, 14, 15);
}

// B-fragment (32x16 bf16): lane n=lane&15, kb=(lane>>4)*8.
// With B transposed in LDS the lane's 16 values are K = [kb..kb+7] and
// [16+kb..16+kb+7] -> 2x ds_load_b128.
__device__ inline v16bf load_b_frag(const __bf16* Bs, int lane) {
  int n = lane & 15, kb = (lane >> 4) * 8;
  v8bf r0 = *(const v8bf*)(Bs + n * LDBT + kb);
  v8bf r1 = *(const v8bf*)(Bs + n * LDBT + 16 + kb);
  return __builtin_shufflevector(r0, r1, 0, 1, 2, 3, 4, 5, 6, 7,
                                 8, 9, 10, 11, 12, 13, 14, 15);
}

// pack two fp32 -> two bf16, single 32-bit LDS store (k must be even)
__device__ inline void store_pk(__bf16* dst, float a, float b) {
  v2bf p; p[0] = (__bf16)a; p[1] = (__bf16)b;
  *(v2bf*)dst = p;
}

// ---------------------------------------------------------------------------
// Implicit-GEMM conv (NCHW), fp32 in/out, bf16 WMMA inner product.
// M = Cout, N = Hout*Wout (per batch via blockIdx.z), K = Cin*ks*ks.
// Weight tile is DMA'd by the Tensor Data Mover into LDS (fp32), then
// pack-converted to the bf16 WMMA tile after threads gather the im2col tile.
// ---------------------------------------------------------------------------
__global__ __launch_bounds__(256) void k_conv_igemm(
    const float* __restrict__ x, const float* __restrict__ w,
    const float* __restrict__ bias, float* __restrict__ y,
    int Cin, int Cout, int H, int W, int Hout, int Wout,
    int ks, int stride, int pad) {
  __shared__ __bf16 As[BM * LDA];
  __shared__ __bf16 Bs[BN * LDBT];
#if USE_TDM
  __shared__ float Stage[BM * BK];
#endif

  const int N    = Hout * Wout;
  const int Ktot = Cin * ks * ks;
  const int n0   = blockIdx.x * BN;
  const int m0   = blockIdx.y * BM;
  const int b    = blockIdx.z;
  const int tid  = threadIdx.x;
  const int lane = tid & 31;
  const int wave = tid >> 5;
  const int wm   = wave >> 1;  // 0..3  (16-row M subtile)
  const int wn   = wave & 1;   // 0..1  (32-col N subtile)

  v8f acc0, acc1;
#pragma unroll
  for (int i = 0; i < 8; ++i) { acc0[i] = 0.f; acc1[i] = 0.f; }

  for (int k0 = 0; k0 < Ktot; k0 += BK) {
#if USE_TDM
    // ---- TDM: DMA weight tile [BM x BK] fp32 -> Stage ----
    if (tid < 32) {
      unsigned long long ga =
          (unsigned long long)(const void*)(w + (size_t)m0 * Ktot + k0);
      unsigned lds = (unsigned)(unsigned long long)(const void*)Stage;
      int td0 = Ktot - k0;           // remaining cols (OOB -> 0)
      int td1 = Cout - m0;           // remaining rows (OOB -> 0)
      ui4 g0;
      g0[0] = 1u;                                        // count=1
      g0[1] = lds;                                       // lds_addr
      g0[2] = (unsigned)(ga & 0xffffffffu);              // global_addr lo
      g0[3] = (unsigned)((ga >> 32) & 0x1ffffffu) | 0x80000000u;  // hi | type=2
      i8v g1;
      g1[0] = 0x00020000;                                // data_size=2 (4B)
      g1[1] = (int)(((unsigned)td0 & 0xffffu) << 16);    // tensor_dim0 lo16
      g1[2] = (int)((((unsigned)td0 >> 16) & 0xffffu) |
                    (((unsigned)td1 & 0xffffu) << 16));  // dim0 hi | dim1 lo
      g1[3] = (int)((((unsigned)td1 >> 16) & 0xffffu) |
                    ((unsigned)BK << 16));               // dim1 hi | tile_dim0
      g1[4] = BM;                                        // tile_dim1
      g1[5] = Ktot;                                      // dim0_stride lo32
      g1[6] = 0;
      g1[7] = 0;
      i4v z4; z4[0] = 0; z4[1] = 0; z4[2] = 0; z4[3] = 0;
      i8v z8;
#pragma unroll
      for (int q = 0; q < 8; ++q) z8[q] = 0;
      __builtin_amdgcn_tensor_load_to_lds(g0, g1, z4, z4, z8, 0);
      __builtin_amdgcn_s_wait_tensorcnt(0);
    }
#else
    // fallback: scalar weight staging, packed pairs
    for (int i = tid; i < BM * BK / 2; i += 256) {
      int idx = 2 * i, m = idx / BK, k = idx % BK;
      int mg = m0 + m;
      float va = (mg < Cout && k0 + k < Ktot) ? w[(size_t)mg * Ktot + k0 + k] : 0.f;
      float vb = (mg < Cout && k0 + k + 1 < Ktot) ? w[(size_t)mg * Ktot + k0 + k + 1] : 0.f;
      store_pk(&As[m * LDA + k], va, vb);
    }
#endif
    // ---- im2col tile, transposed [BN][BK], packed pair stores ----
    for (int i = tid; i < BN * BK / 2; i += 256) {
      int idx = 2 * i, n = idx / BK, k = idx % BK;
      int ng = n0 + n;
      float va = 0.f, vb = 0.f;
      if (ng < N) {
        int oy = ng / Wout, ox = ng % Wout;
#pragma unroll
        for (int t = 0; t < 2; ++t) {
          int kg = k0 + k + t;
          if (kg < Ktot) {
            int ci = kg / (ks * ks);
            int r  = kg % (ks * ks);
            int iy = oy * stride - pad + r / ks;
            int ix = ox * stride - pad + r % ks;
            if (iy >= 0 && iy < H && ix >= 0 && ix < W) {
              float v = x[(((size_t)b * Cin + ci) * H + iy) * W + ix];
              if (t == 0) va = v; else vb = v;
            }
          }
        }
      }
      store_pk(&Bs[n * LDBT + k], va, vb);
    }
#if USE_TDM
    __syncthreads();  // TDM data + Bs visible
    // pack-convert fp32 Stage -> bf16 A tile
    for (int i = tid; i < BM * BK / 2; i += 256) {
      int idx = 2 * i, m = idx / BK, k = idx % BK;
      float2 v = *(const float2*)(Stage + m * BK + k);
      store_pk(&As[m * LDA + k], v.x, v.y);
    }
#endif
    __syncthreads();

    v16bf a  = load_a_frag(As + wm * 16 * LDA, lane);
    v16bf b0 = load_b_frag(Bs + (wn * 32) * LDBT, lane);
    v16bf b1 = load_b_frag(Bs + (wn * 32 + 16) * LDBT, lane);
    acc0 = wmma_bf16(a, b0, acc0);
    acc1 = wmma_bf16(a, b1, acc1);
    __syncthreads();
  }

  // store: VGPR r -> M = r (+8 for lanes>=16), N = lane&15
  const int nn   = lane & 15;
  const int mofs = (lane >> 4) * 8;
#pragma unroll
  for (int r = 0; r < 8; ++r) {
    int mg = m0 + wm * 16 + mofs + r;
    if (mg >= Cout) continue;
    float bb = bias[mg];
    int ng0 = n0 + wn * 32 + nn;
    if (ng0 < N)      y[((size_t)b * Cout + mg) * N + ng0]      = acc0[r] + bb;
    if (ng0 + 16 < N) y[((size_t)b * Cout + mg) * N + ng0 + 16] = acc1[r] + bb;
  }
}

// ---------------------------------------------------------------------------
// Generic strided WMMA GEMM: C[M,N] = alpha * A * B, C row-major.
// A(m,k) = A[m*sa_m + k*sa_k];  B(k,n) = B[k*sb_k + n*sb_n]
// ---------------------------------------------------------------------------
__global__ __launch_bounds__(256) void k_gemm(
    const float* __restrict__ A, const float* __restrict__ B,
    float* __restrict__ C, int M, int N, int K,
    int sa_m, int sa_k, int sb_k, int sb_n, float alpha) {
  __shared__ __bf16 As[BM * LDA];
  __shared__ __bf16 Bs[BN * LDBT];

  const int n0   = blockIdx.x * BN;
  const int m0   = blockIdx.y * BM;
  const int tid  = threadIdx.x;
  const int lane = tid & 31;
  const int wave = tid >> 5;
  const int wm   = wave >> 1;
  const int wn   = wave & 1;

  v8f acc0, acc1;
#pragma unroll
  for (int i = 0; i < 8; ++i) { acc0[i] = 0.f; acc1[i] = 0.f; }

  for (int k0 = 0; k0 < K; k0 += BK) {
    for (int i = tid; i < BM * BK / 2; i += 256) {
      int idx = 2 * i, m = idx / BK, k = idx % BK;
      int mg = m0 + m;
      float va = (mg < M && k0 + k < K)
                     ? A[(size_t)mg * sa_m + (size_t)(k0 + k) * sa_k] : 0.f;
      float vb = (mg < M && k0 + k + 1 < K)
                     ? A[(size_t)mg * sa_m + (size_t)(k0 + k + 1) * sa_k] : 0.f;
      store_pk(&As[m * LDA + k], va, vb);
    }
    // prefetch next A k-panel (global_prefetch_b8 path)
    if (k0 + BK < K && tid < BM) {
      int mg = m0 + tid;
      if (mg < M) __builtin_prefetch(&A[(size_t)mg * sa_m + (size_t)(k0 + BK) * sa_k], 0, 1);
    }
    for (int i = tid; i < BN * BK / 2; i += 256) {
      int idx = 2 * i, n = idx / BK, k = idx % BK;
      int ng = n0 + n;
      float va = (ng < N && k0 + k < K)
                     ? B[(size_t)(k0 + k) * sb_k + (size_t)ng * sb_n] : 0.f;
      float vb = (ng < N && k0 + k + 1 < K)
                     ? B[(size_t)(k0 + k + 1) * sb_k + (size_t)ng * sb_n] : 0.f;
      store_pk(&Bs[n * LDBT + k], va, vb);
    }
    __syncthreads();
    v16bf a  = load_a_frag(As + wm * 16 * LDA, lane);
    v16bf b0 = load_b_frag(Bs + (wn * 32) * LDBT, lane);
    v16bf b1 = load_b_frag(Bs + (wn * 32 + 16) * LDBT, lane);
    acc0 = wmma_bf16(a, b0, acc0);
    acc1 = wmma_bf16(a, b1, acc1);
    __syncthreads();
  }

  const int nn   = lane & 15;
  const int mofs = (lane >> 4) * 8;
#pragma unroll
  for (int r = 0; r < 8; ++r) {
    int mg = m0 + wm * 16 + mofs + r;
    if (mg >= M) continue;
    int ng0 = n0 + wn * 32 + nn;
    if (ng0 < N)      C[(size_t)mg * N + ng0]      = alpha * acc0[r];
    if (ng0 + 16 < N) C[(size_t)mg * N + ng0 + 16] = alpha * acc1[r];
  }
}

// ---------------------------------------------------------------------------
// Fused GroupNorm (32 groups) + optional SiLU.  grid = (32, B)
// ---------------------------------------------------------------------------
__global__ __launch_bounds__(256) void k_groupnorm(
    const float* __restrict__ x, float* __restrict__ y,
    const float* __restrict__ g, const float* __restrict__ be,
    int C, int HW, int cpg, int do_silu) {
  const int gb = blockIdx.y, gr = blockIdx.x;
  const float* xp = x + ((size_t)gb * C + (size_t)gr * cpg) * HW;
  float* yp = y + ((size_t)gb * C + (size_t)gr * cpg) * HW;
  const long n = (long)cpg * HW;

  float s = 0.f, s2 = 0.f;
  for (long i = threadIdx.x; i < n; i += blockDim.x) {
    float v = xp[i];
    s += v; s2 += v * v;
  }
  __shared__ float red[16];
  int ln = threadIdx.x & 31, wv = threadIdx.x >> 5;
  for (int o = 16; o; o >>= 1) { s += __shfl_xor(s, o); s2 += __shfl_xor(s2, o); }
  if (ln == 0) { red[wv] = s; red[8 + wv] = s2; }
  __syncthreads();
  if (threadIdx.x == 0) {
    float ts = 0.f, ts2 = 0.f;
    for (int i = 0; i < 8; ++i) { ts += red[i]; ts2 += red[8 + i]; }
    red[0] = ts; red[8] = ts2;
  }
  __syncthreads();
  float mean = red[0] / (float)n;
  float var  = red[8] / (float)n - mean * mean;
  float inv  = rsqrtf(var + 1e-6f);

  for (long i = threadIdx.x; i < n; i += blockDim.x) {
    int c = gr * cpg + (int)(i / HW);
    float v = (xp[i] - mean) * inv * g[c] + be[c];
    if (do_silu) v = v / (1.f + __expf(-v));
    yp[i] = v;
  }
}

// row softmax, cols <= 256, one block per row
__global__ __launch_bounds__(256) void k_softmax(float* a, int cols) {
  float* p = a + (size_t)blockIdx.x * cols;
  int tid = threadIdx.x;
  __shared__ float sm[256];
  float v = (tid < cols) ? p[tid] : -1e30f;
  sm[tid] = v; __syncthreads();
  for (int o = 128; o; o >>= 1) { if (tid < o) sm[tid] = fmaxf(sm[tid], sm[tid + o]); __syncthreads(); }
  float mx = sm[0]; __syncthreads();
  float e = (tid < cols) ? __expf(v - mx) : 0.f;
  sm[tid] = e; __syncthreads();
  for (int o = 128; o; o >>= 1) { if (tid < o) sm[tid] += sm[tid + o]; __syncthreads(); }
  if (tid < cols) p[tid] = e / sm[0];
}

// ---------------------------------------------------------------------------
// VQ helpers
// ---------------------------------------------------------------------------
__global__ void k_enorm(const float* __restrict__ emb, float* __restrict__ en, int V, int C) {
  int j = blockIdx.x * blockDim.x + threadIdx.x;
  if (j >= V) return;
  float s = 0.f;
  for (int c = 0; c < C; ++c) { float v = emb[(size_t)j * C + c]; s += v * v; }
  en[j] = s;
}

__global__ __launch_bounds__(256) void k_argmin(
    const float* __restrict__ dots, const float* __restrict__ en,
    int V, int* __restrict__ idx) {
  int row = blockIdx.x, tid = threadIdx.x;
  const float* dp = dots + (size_t)row * V;
  float best = 1e30f; int bi = 0;
  for (int j = tid; j < V; j += 256) {
    float d = en[j] - 2.f * dp[j];
    if (d < best) { best = d; bi = j; }
  }
  __shared__ float sv[256]; __shared__ int sj[256];
  sv[tid] = best; sj[tid] = bi; __syncthreads();
  for (int o = 128; o; o >>= 1) {
    if (tid < o) {
      if (sv[tid + o] < sv[tid] ||
          (sv[tid + o] == sv[tid] && sj[tid + o] < sj[tid])) {
        sv[tid] = sv[tid + o]; sj[tid] = sj[tid + o];
      }
    }
    __syncthreads();
  }
  if (tid == 0) idx[row] = sj[0];
}

__global__ void k_gather_emb(const float* __restrict__ emb, const int* __restrict__ idx,
                             float* __restrict__ out, int B, int C, int PP) {
  int t = blockIdx.x * blockDim.x + threadIdx.x;
  int n = B * C * PP;
  if (t >= n) return;
  int b = t / (C * PP);
  int r = t % (C * PP);
  int c = r / PP, p = r % PP;
  out[t] = emb[(size_t)idx[b * PP + p] * C + c];
}

__device__ inline float cubw(float t) {
  t = fabsf(t);
  if (t <= 1.f) return ((1.5f * t - 2.5f) * t) * t + 1.f;
  if (t < 2.f)  return ((-0.5f * t + 2.5f) * t - 4.f) * t + 2.f;
  return 0.f;
}

// bicubic (Keys a=-0.5, half-pixel) pn x pn -> 16 x 16; identity at pn==16
__global__ void k_resize_cubic(const float* __restrict__ in, float* __restrict__ out,
                               int B, int C, int pn) {
  int t = blockIdx.x * blockDim.x + threadIdx.x;
  int n = B * C * 256;
  if (t >= n) return;
  int ox = t % 16, oy = (t / 16) % 16, bc = t / 256;
  float sc = (float)pn / 16.f;
  float sx = (ox + 0.5f) * sc - 0.5f;
  float sy = (oy + 0.5f) * sc - 0.5f;
  int ix = (int)floorf(sx), iy = (int)floorf(sy);
  float fx = sx - ix, fy = sy - iy;
  float acc = 0.f;
#pragma unroll
  for (int dy = -1; dy <= 2; ++dy) {
    int yy = iy + dy; yy = yy < 0 ? 0 : (yy > pn - 1 ? pn - 1 : yy);
    float wy = cubw(fy - dy);
#pragma unroll
    for (int dx = -1; dx <= 2; ++dx) {
      int xx = ix + dx; xx = xx < 0 ? 0 : (xx > pn - 1 ? pn - 1 : xx);
      acc += wy * cubw(fx - dx) * in[((size_t)bc * pn + yy) * pn + xx];
    }
  }
  out[t] = acc;
}

__global__ void k_avgpool(const float* __restrict__ in, float* __restrict__ out,
                          int B, int C, int H, int pn) {
  int t = blockIdx.x * blockDim.x + threadIdx.x;
  int n = B * C * pn * pn;
  if (t >= n) return;
  int px = t % pn, py = (t / pn) % pn, bc = t / (pn * pn);
  int f = H / pn;
  float s = 0.f;
  for (int dy = 0; dy < f; ++dy)
    for (int dx = 0; dx < f; ++dx)
      s += in[((size_t)bc * H + py * f + dy) * H + px * f + dx];
  out[t] = s / (float)(f * f);
}

// ---------------------------------------------------------------------------
// Elementwise utility kernels
// ---------------------------------------------------------------------------
__global__ void k_add(const float* a, const float* b, float* o, int n) {
  int i = blockIdx.x * blockDim.x + threadIdx.x; if (i < n) o[i] = a[i] + b[i];
}
__global__ void k_addip(float* a, const float* b, int n) {
  int i = blockIdx.x * blockDim.x + threadIdx.x; if (i < n) a[i] += b[i];
}
__global__ void k_subip(float* a, const float* b, int n) {
  int i = blockIdx.x * blockDim.x + threadIdx.x; if (i < n) a[i] -= b[i];
}
__global__ void k_lerp_half(float* a, const float* b, int n) {
  int i = blockIdx.x * blockDim.x + threadIdx.x; if (i < n) a[i] = 0.5f * (a[i] + b[i]);
}
__global__ void k_fill(float* a, float v, int n) {
  int i = blockIdx.x * blockDim.x + threadIdx.x; if (i < n) a[i] = v;
}
__global__ void k_clip(const float* a, float* o, int n) {
  int i = blockIdx.x * blockDim.x + threadIdx.x;
  if (i < n) o[i] = fminf(1.f, fmaxf(-1.f, a[i]));
}
__global__ void k_upsample2x(const float* __restrict__ in, float* __restrict__ out,
                             int B, int C, int H, int W) {
  int t = blockIdx.x * blockDim.x + threadIdx.x;
  int n = B * C * 4 * H * W;
  if (t >= n) return;
  int W2 = 2 * W, H2 = 2 * H;
  int ox = t % W2, oy = (t / W2) % H2, bc = t / (W2 * H2);
  out[t] = in[((size_t)bc * H + (oy >> 1)) * W + (ox >> 1)];
}
__global__ __launch_bounds__(256) void k_mse_accum(
    const float* a, const float* b, int n, float coef, float* loss) {
  float s = 0.f;
  for (int i = blockIdx.x * blockDim.x + threadIdx.x; i < n; i += gridDim.x * blockDim.x) {
    float d = a[i] - b[i]; s += d * d;
  }
  for (int o = 16; o; o >>= 1) s += __shfl_xor(s, o);
  __shared__ float red[8];
  int ln = threadIdx.x & 31, wv = threadIdx.x >> 5;
  if (ln == 0) red[wv] = s;
  __syncthreads();
  if (threadIdx.x == 0) {
    float ts = 0.f;
    for (int i = 0; i < 8; ++i) ts += red[i];
    atomicAdd(loss, ts * coef);
  }
}

// ---------------------------------------------------------------------------
// Host-side parameter walking (JAX pytree flatten order: dict keys sorted,
// lists positional, None dropped)
// ---------------------------------------------------------------------------
struct Cursor { void* const* in; int i; const float* next() { return (const float*)in[i++]; } };
struct ConvP { const float* b; const float* w; };
struct GnP   { const float* b; const float* g; };
struct ResP  { ConvP c1, c2; GnP n1, n2; ConvP nin; bool has_nin; };
struct AttnP { GnP n; ConvP proj, qkv; };

static ConvP take_conv(Cursor& c) { ConvP p; p.b = c.next(); p.w = c.next(); return p; }
static GnP   take_gn(Cursor& c)   { GnP p;  p.b = c.next(); p.g = c.next(); return p; }
static ResP  take_res(Cursor& c, bool nin) {
  ResP r; r.c1 = take_conv(c); r.c2 = take_conv(c);
  r.n1 = take_gn(c); r.n2 = take_gn(c);
  r.has_nin = nin; if (nin) r.nin = take_conv(c);
  return r;
}
static AttnP take_attn(Cursor& c) {
  AttnP a; a.n = take_gn(c); a.proj = take_conv(c); a.qkv = take_conv(c); return a;
}

static inline int cdiv(int a, int b) { return (a + b - 1) / b; }

static void conv2d(hipStream_t st, const float* x, const ConvP& p, float* y,
                   int B, int Cin, int Cout, int H, int W,
                   int ks, int stride, int pad, int Ho, int Wo) {
  dim3 g(cdiv(Ho * Wo, BN), cdiv(Cout, BM), B);
  k_conv_igemm<<<g, 256, 0, st>>>(x, p.w, p.b, y, Cin, Cout, H, W, Ho, Wo, ks, stride, pad);
}
static void gemm(hipStream_t st, const float* A, const float* Bp, float* C,
                 int M, int N, int K, int sam, int sak, int sbk, int sbn, float alpha) {
  dim3 g(cdiv(N, BN), cdiv(M, BM), 1);
  k_gemm<<<g, 256, 0, st>>>(A, Bp, C, M, N, K, sam, sak, sbk, sbn, alpha);
}
static void gnorm(hipStream_t st, const float* x, float* y, const GnP& p,
                  int B, int C, int HW, int silu) {
  k_groupnorm<<<dim3(32, B), 256, 0, st>>>(x, y, p.g, p.b, C, HW, C / 32, silu);
}
static void eadd(hipStream_t st, const float* a, const float* b, float* o, int n) {
  k_add<<<cdiv(n, 256), 256, 0, st>>>(a, b, o, n);
}

static void resblock(hipStream_t st, const float* x, float* out, float* t1, float* t2,
                     const ResP& r, int B, int Cin, int Cout, int H, int W) {
  int HW = H * W;
  gnorm(st, x, t1, r.n1, B, Cin, HW, 1);
  conv2d(st, t1, r.c1, t2, B, Cin, Cout, H, W, 3, 1, 1, H, W);
  gnorm(st, t2, t1, r.n2, B, Cout, HW, 1);
  conv2d(st, t1, r.c2, t2, B, Cout, Cout, H, W, 3, 1, 1, H, W);
  if (r.has_nin) {
    conv2d(st, x, r.nin, t1, B, Cin, Cout, H, W, 1, 1, 0, H, W);
    eadd(st, t1, t2, out, B * Cout * HW);
  } else {
    eadd(st, x, t2, out, B * Cout * HW);
  }
}

static void attnblock(hipStream_t st, const float* x, float* out, float* t1, float* t2,
                      float* qkv, float* sc, const AttnP& p, int B, int C, int HW) {
  gnorm(st, x, t1, p.n, B, C, HW, 0);
  conv2d(st, t1, p.qkv, qkv, B, C, 3 * C, HW, 1, 1, 1, 0, HW, 1);
  float scl = rsqrtf((float)C);
  for (int b = 0; b < B; ++b) {
    const float* q = qkv + (size_t)b * 3 * C * HW;
    const float* k = q + (size_t)C * HW;
    // scores[n][m] = sum_c q[c][n]*k[c][m]  (softmax over m)
    gemm(st, q, k, sc + (size_t)b * HW * HW, HW, HW, C, 1, HW, HW, 1, scl);
  }
  k_softmax<<<B * HW, 256, 0, st>>>(sc, HW);
  for (int b = 0; b < B; ++b) {
    const float* q = qkv + (size_t)b * 3 * C * HW;
    const float* v = q + (size_t)2 * C * HW;
    // h[c][n] = sum_m v[c][m]*w[n][m]
    gemm(st, v, sc + (size_t)b * HW * HW, t2 + (size_t)b * C * HW,
         C, HW, HW, HW, 1, 1, HW, 1.f);
  }
  conv2d(st, t2, p.proj, t1, B, C, C, HW, 1, 1, 1, 0, HW, 1);
  eadd(st, x, t1, out, B * C * HW);
}

// ---------------------------------------------------------------------------
// Full VQVAE forward
// ---------------------------------------------------------------------------
extern "C" void kernel_launch(void* const* d_in, const int* in_sizes, int n_in,
                              void* d_out, int out_size, void* d_ws, size_t ws_size,
                              hipStream_t stream) {
  (void)in_sizes; (void)n_in; (void)out_size; (void)ws_size;
  const int B = 2, ZC = 32, VOCAB = 4096;
  const int cm[5] = {1, 1, 2, 2, 4};

  Cursor cur{d_in, 0};
  const float* inp = cur.next();  // [2,3,256,256]

  // ---- dec (keys: conv_in, conv_out, mid{attn,b1,b2}, norm_out, up[0..4]) ----
  ConvP dec_cin  = take_conv(cur);
  ConvP dec_cout = take_conv(cur);
  AttnP dmid_at  = take_attn(cur);
  ResP  dmid_b1  = take_res(cur, false);
  ResP  dmid_b2  = take_res(cur, false);
  GnP   dec_no   = take_gn(cur);
  struct DecLvl { AttnP at[3]; int n_at; ResP blk[3]; ConvP up; bool has_up; } dl[5];
  dl[0].n_at = 0;
  for (int j = 0; j < 3; ++j) dl[0].blk[j] = take_res(cur, false);
  dl[0].has_up = false;
  dl[1].n_at = 0;
  dl[1].blk[0] = take_res(cur, true);
  dl[1].blk[1] = take_res(cur, false);
  dl[1].blk[2] = take_res(cur, false);
  dl[1].up = take_conv(cur); dl[1].has_up = true;
  dl[2].n_at = 0;
  for (int j = 0; j < 3; ++j) dl[2].blk[j] = take_res(cur, false);
  dl[2].up = take_conv(cur); dl[2].has_up = true;
  dl[3].n_at = 0;
  dl[3].blk[0] = take_res(cur, true);
  dl[3].blk[1] = take_res(cur, false);
  dl[3].blk[2] = take_res(cur, false);
  dl[3].up = take_conv(cur); dl[3].has_up = true;
  dl[4].n_at = 3;
  for (int j = 0; j < 3; ++j) dl[4].at[j] = take_attn(cur);
  for (int j = 0; j < 3; ++j) dl[4].blk[j] = take_res(cur, false);
  dl[4].up = take_conv(cur); dl[4].has_up = true;

  // ---- emb ----
  const float* emb = cur.next();  // [4096,32]

  // ---- enc (keys: conv_in, conv_out, down[0..4], mid{attn,b1,b2}, norm_out) ----
  ConvP enc_cin  = take_conv(cur);
  ConvP enc_cout = take_conv(cur);
  struct EncLvl { AttnP at[2]; int n_at; ResP blk[2]; ConvP down; bool has_down; } el[5];
  for (int i = 0; i < 5; ++i) {
    el[i].n_at = (i == 4) ? 2 : 0;
    for (int j = 0; j < el[i].n_at; ++j) el[i].at[j] = take_attn(cur);
    bool nin0 = (i == 2 || i == 4);  // 64->128 and 128->256
    el[i].blk[0] = take_res(cur, nin0);
    el[i].blk[1] = take_res(cur, false);
    el[i].has_down = (i != 4);
    if (el[i].has_down) el[i].down = take_conv(cur);
  }
  AttnP emid_at = take_attn(cur);
  ResP  emid_b1 = take_res(cur, false);
  ResP  emid_b2 = take_res(cur, false);
  GnP   enc_no  = take_gn(cur);

  // ---- phis, post_quant, quant ----
  ConvP phis[4];
  for (int i = 0; i < 4; ++i) phis[i] = take_conv(cur);
  ConvP post_q = take_conv(cur);
  ConvP quant  = take_conv(cur);

  // ---- workspace arena ----
  char* w8 = (char*)d_ws;
  const size_t SLOT = (size_t)34 << 20;
  float* S[6];
  for (int i = 0; i < 6; ++i) S[i] = (float*)(w8 + i * SLOT);
  char* sm = w8 + 6 * SLOT;
  float* F     = (float*)sm;            sm += 1 << 20;
  float* FHAT  = (float*)sm;            sm += 1 << 20;
  float* FREST = (float*)sm;            sm += 1 << 20;
  float* Z     = (float*)sm;            sm += 1 << 20;
  float* HS    = (float*)sm;            sm += 1 << 20;
  float* HT    = (float*)sm;            sm += 1 << 20;
  float* EN    = (float*)sm;            sm += 1 << 20;
  int*   IDX   = (int*)sm;              sm += 1 << 20;
  float* DOTS  = (float*)sm;            // up to B*256*4096*4 = 8MB

  // ================= ENCODER =================
  int H = 256, W = 256, C = 64;
  float* a = S[0];
  float* b2 = S[3];
  conv2d(stream, inp, enc_cin, a, B, 3, 64, H, W, 3, 1, 1, H, W);
  for (int i = 0; i < 5; ++i) {
    int bo = 64 * cm[i];
    for (int j = 0; j < 2; ++j) {
      resblock(stream, a, b2, S[1], S[2], el[i].blk[j], B, C, bo, H, W);
      { float* t = a; a = b2; b2 = t; }
      C = bo;
      if (el[i].n_at) {
        attnblock(stream, a, b2, S[1], S[2], S[4], S[5], el[i].at[j], B, C, H * W);
        { float* t = a; a = b2; b2 = t; }
      }
    }
    if (el[i].has_down) {
      conv2d(stream, a, el[i].down, b2, B, C, C, H, W, 3, 2, 0, H / 2, W / 2);
      { float* t = a; a = b2; b2 = t; }
      H /= 2; W /= 2;
    }
  }
  resblock(stream, a, b2, S[1], S[2], emid_b1, B, C, C, H, W);
  { float* t = a; a = b2; b2 = t; }
  attnblock(stream, a, b2, S[1], S[2], S[4], S[5], emid_at, B, C, H * W);
  { float* t = a; a = b2; b2 = t; }
  resblock(stream, a, b2, S[1], S[2], emid_b2, B, C, C, H, W);
  { float* t = a; a = b2; b2 = t; }
  gnorm(stream, a, S[1], enc_no, B, C, H * W, 1);
  conv2d(stream, S[1], enc_cout, S[2], B, C, ZC, H, W, 3, 1, 1, H, W);
  conv2d(stream, S[2], quant, F, B, ZC, ZC, H, W, 3, 1, 1, H, W);

  // ================= MULTISCALE VQ =================
  const int FN = B * ZC * 16 * 16;  // 16384
  float* outp  = (float*)d_out;
  float* lossp = outp + (size_t)B * 3 * 256 * 256;
  k_fill<<<cdiv(FN, 256), 256, 0, stream>>>(FHAT, 0.f, FN);
  k_fill<<<1, 256, 0, stream>>>(lossp, 0.f, 1);
  (void)hipMemcpyAsync(FREST, F, (size_t)FN * sizeof(float), hipMemcpyDeviceToDevice, stream);
  k_enorm<<<cdiv(VOCAB, 256), 256, 0, stream>>>(emb, EN, VOCAB, ZC);

  const int   pns[5]  = {1, 2, 4, 8, 16};
  const int   phii[5] = {0, 1, 1, 2, 3};
  const float coef    = (1.f + 0.25f) / (5.f * (float)FN);
  for (int si = 0; si < 5; ++si) {
    int pn = pns[si], PP = pn * pn;
    const float* zsrc;
    if (si < 4) {
      k_avgpool<<<cdiv(B * ZC * PP, 256), 256, 0, stream>>>(FREST, Z, B, ZC, 16, pn);
      zsrc = Z;
    } else {
      zsrc = FREST;
    }
    for (int bb = 0; bb < B; ++bb) {
      gemm(stream, zsrc + (size_t)bb * ZC * PP, emb, DOTS + (size_t)bb * PP * VOCAB,
           PP, VOCAB, ZC, 1, PP, 1, ZC, 1.f);
    }
    k_argmin<<<B * PP, 256, 0, stream>>>(DOTS, EN, VOCAB, IDX);
    k_gather_emb<<<cdiv(B * ZC * PP, 256), 256, 0, stream>>>(emb, IDX, HT, B, ZC, PP);
    k_resize_cubic<<<cdiv(B * ZC * 256, 256), 256, 0, stream>>>(HT, HS, B, ZC, pn);
    conv2d(stream, HS, phis[phii[si]], HT, B, ZC, ZC, 16, 16, 3, 1, 1, 16, 16);
    k_lerp_half<<<cdiv(FN, 256), 256, 0, stream>>>(HS, HT, FN);
    k_addip<<<cdiv(FN, 256), 256, 0, stream>>>(FHAT, HS, FN);
    k_subip<<<cdiv(FN, 256), 256, 0, stream>>>(FREST, HS, FN);
    k_mse_accum<<<64, 256, 0, stream>>>(FHAT, F, FN, coef, lossp);
  }

  // ================= DECODER =================
  H = 16; W = 16; C = 256;
  conv2d(stream, FHAT, post_q, Z, B, ZC, ZC, H, W, 3, 1, 1, H, W);
  a = S[0]; b2 = S[3];
  conv2d(stream, Z, dec_cin, a, B, ZC, 256, H, W, 3, 1, 1, H, W);
  resblock(stream, a, b2, S[1], S[2], dmid_b1, B, C, C, H, W);
  { float* t = a; a = b2; b2 = t; }
  attnblock(stream, a, b2, S[1], S[2], S[4], S[5], dmid_at, B, C, H * W);
  { float* t = a; a = b2; b2 = t; }
  resblock(stream, a, b2, S[1], S[2], dmid_b2, B, C, C, H, W);
  { float* t = a; a = b2; b2 = t; }
  for (int i = 4; i >= 0; --i) {
    int bo = 64 * cm[i];
    for (int j = 0; j < 3; ++j) {
      resblock(stream, a, b2, S[1], S[2], dl[i].blk[j], B, C, bo, H, W);
      { float* t = a; a = b2; b2 = t; }
      C = bo;
      if (dl[i].n_at) {
        attnblock(stream, a, b2, S[1], S[2], S[4], S[5], dl[i].at[j], B, C, H * W);
        { float* t = a; a = b2; b2 = t; }
      }
    }
    if (dl[i].has_up) {
      k_upsample2x<<<cdiv(B * C * 4 * H * W, 256), 256, 0, stream>>>(a, b2, B, C, H, W);
      { float* t = a; a = b2; b2 = t; }
      H *= 2; W *= 2;
      conv2d(stream, a, dl[i].up, b2, B, C, C, H, W, 3, 1, 1, H, W);
      { float* t = a; a = b2; b2 = t; }
    }
  }
  gnorm(stream, a, S[1], dec_no, B, C, H * W, 1);
  conv2d(stream, S[1], dec_cout, S[2], B, C, 3, H, W, 3, 1, 1, H, W);
  k_clip<<<cdiv(B * 3 * H * W, 256), 256, 0, stream>>>(S[2], outp, B * 3 * H * W);
}
